// CrossAttention_62354335203903
// MI455X (gfx1250) — matrix operand order repaired and checked by reading, not compiled, not run on hardware
//
#include <hip/hip_runtime.h>

// CDNA5 / gfx1250 cross-attention pipeline, bf16 WMMA (v_wmma_f32_16x16x32_bf16)
// + TDM tensor_load_to_lds staging (guarded) + global_prefetch hints.
// Workspace layout (bf16 elements, SZ = B*C*N = 2*256*4096):
//   [0,SZ)      nq      (GroupNorm(input))
//   [SZ,2SZ)    nkv     (GroupNorm(c))
//   [2SZ,3SZ)   q       (wq @ nq, pre-scaled by 1/sqrt(C))
//   [3SZ,5SZ)   kv      (wkv @ nkv; per head h: rows h*128..+63 = k, +64..+127 = v)
//   [5SZ,6SZ)   o       (attention output, channel-major)
// Total 24 MiB of workspace.

#define NTOT 4096   // H*W
#define CCH  256    // channels

typedef __bf16 v16bf __attribute__((ext_vector_type(16)));
typedef float  v8f   __attribute__((ext_vector_type(8)));
typedef unsigned int u32x4 __attribute__((ext_vector_type(4)));
typedef int          i32x4 __attribute__((ext_vector_type(4)));
typedef int          i32x8 __attribute__((ext_vector_type(8)));

#if defined(__has_builtin)
#if __has_builtin(__builtin_amdgcn_tensor_load_to_lds) && __has_builtin(__builtin_amdgcn_s_wait_tensorcnt)
#define USE_TDM 1
#endif
#endif
#ifndef USE_TDM
#define USE_TDM 0
#endif

union Frag16 { v16bf v; unsigned u[8]; unsigned short h[16]; };

__device__ __forceinline__ unsigned short f2bf(float f) {
  union { __bf16 b; unsigned short u; } x;
  x.b = (__bf16)f;                       // native convert (RNE) on gfx1250
  return x.u;
}

// ---------------------------------------------------------------------------
// GroupNorm: one block per (tensor, batch, group). 8 channels * 4096 = 32768.
// ---------------------------------------------------------------------------
__global__ void __launch_bounds__(256)
gn_kernel(const float* __restrict__ x0, const float* __restrict__ x1,
          const float* __restrict__ gamma, const float* __restrict__ beta,
          unsigned short* __restrict__ y0, unsigned short* __restrict__ y1)
{
  const float* x = blockIdx.y ? x1 : x0;
  unsigned short* y = blockIdx.y ? y1 : y0;
  const int b = blockIdx.x >> 5;
  const int g = blockIdx.x & 31;
  const long base = ((long)b * CCH + (long)g * 8) * NTOT;
  const float4* xv = (const float4*)(x + base);
  const int tid = threadIdx.x;

  float s = 0.f, ss = 0.f;
  #pragma unroll 4
  for (int i = 0; i < 32; ++i) {
    float4 v = xv[tid + i * 256];
    s  += v.x + v.y + v.z + v.w;
    ss += v.x * v.x + v.y * v.y + v.z * v.z + v.w * v.w;
  }
  __shared__ float rs[256], rq[256];
  rs[tid] = s; rq[tid] = ss;
  __syncthreads();
  for (int o = 128; o > 0; o >>= 1) {
    if (tid < o) { rs[tid] += rs[tid + o]; rq[tid] += rq[tid + o]; }
    __syncthreads();
  }
  const float mean = rs[0] * (1.0f / 32768.0f);
  const float var  = rq[0] * (1.0f / 32768.0f) - mean * mean;
  const float rstd = rsqrtf(var + 1e-5f);

  for (int i = 0; i < 32; ++i) {
    int idx4 = tid + i * 256;                 // float4 index within group
    int c = g * 8 + (idx4 >> 10);             // 1024 float4 per channel
    float ga = gamma[c] * rstd;
    float be = beta[c] - mean * ga;
    float4 v = xv[idx4];
    unsigned short* yp = y + base + (long)idx4 * 4;
    yp[0] = f2bf(v.x * ga + be); yp[1] = f2bf(v.y * ga + be);
    yp[2] = f2bf(v.z * ga + be); yp[3] = f2bf(v.w * ga + be);
  }
}

// ---------------------------------------------------------------------------
// Generic GEMM: out[m,n] = scale * sum_c W[m,c] * X[c,n] (+bias)(+residual)
// W fp32 [Odim x 256], X bf16 [256 x 4096] per batch (blockIdx.z).
// Block tile 128x128, 8 waves, each wave 32x64 (2x4 WMMA accumulators).
// ---------------------------------------------------------------------------
__global__ void __launch_bounds__(256)
gemm_bf16(const float* __restrict__ W, const unsigned short* __restrict__ X,
          unsigned short* __restrict__ outb, float* __restrict__ outf,
          const float* __restrict__ bias, const float* __restrict__ resid,
          int Odim, float scale)
{
  const int bN = blockIdx.x * 128;
  const int bM = blockIdx.y * 128;
  const int bb = blockIdx.z;
  X += (long)bb * CCH * NTOT;

  const int tid  = threadIdx.x;
  const int lane = tid & 31;
  const int w    = tid >> 5;
  const int hi   = lane >> 4;
  const int l16  = lane & 15;

  __shared__ unsigned short lA[128 * 34];   // [m][k], padded row stride 34
  __shared__ unsigned short lB[128 * 34];   // [n][k] (transposed), stride 34
  unsigned* lAu = (unsigned*)lA;
  unsigned* lBu = (unsigned*)lB;

  const int wm = (w >> 1) * 32;
  const int wn = (w & 1) * 64;

  v8f zero = {0.f, 0.f, 0.f, 0.f, 0.f, 0.f, 0.f, 0.f};
  v8f acc[2][4];
  #pragma unroll
  for (int mt = 0; mt < 2; ++mt)
    #pragma unroll
    for (int nt = 0; nt < 4; ++nt) acc[mt][nt] = zero;

  for (int k0 = 0; k0 < CCH; k0 += 32) {
    __syncthreads();
    {   // stage A: W rows bM..+127, cols k0..+31 (fp32 -> bf16)
      int r = tid >> 1, half = (tid & 1) * 16;
      const float* wp = W + (long)(bM + r) * CCH + k0 + half;
      unsigned short* dp = lA + r * 34 + half;
      #pragma unroll
      for (int j = 0; j < 16; j += 4) {
        float4 v = *(const float4*)(wp + j);
        dp[j] = f2bf(v.x); dp[j + 1] = f2bf(v.y);
        dp[j + 2] = f2bf(v.z); dp[j + 3] = f2bf(v.w);
      }
      if (k0 + 32 < CCH) __builtin_prefetch(wp + 32, 0, 3);   // global_prefetch
    }
    {   // stage B transposed: X rows k0..+31, cols bN..+127 -> lB[n][k]
      int k = tid >> 3, seg = (tid & 7) * 16;
      const unsigned short* xp = X + (long)(k0 + k) * NTOT + bN + seg;
      #pragma unroll
      for (int j = 0; j < 16; ++j) lB[(seg + j) * 34 + k] = xp[j];
      if (k0 + 32 < CCH) __builtin_prefetch(xp + 32 * NTOT, 0, 3);
    }
    __syncthreads();

    Frag16 a[2], bf[4];
    #pragma unroll
    for (int mt = 0; mt < 2; ++mt) {
      int row = wm + mt * 16 + l16;
      #pragma unroll
      for (int j = 0; j < 8; ++j)
        a[mt].u[j] = lAu[row * 17 + j + 8 * hi];
    }
    #pragma unroll
    for (int nt = 0; nt < 4; ++nt) {
      int row = wn + nt * 16 + l16;
      #pragma unroll
      for (int j = 0; j < 8; ++j)
        bf[nt].u[j] = lBu[row * 17 + j + 8 * hi];
    }
    #pragma unroll
    for (int mt = 0; mt < 2; ++mt)
      #pragma unroll
      for (int nt = 0; nt < 4; ++nt)
        acc[mt][nt] = __builtin_amdgcn_wmma_f32_16x16x32_bf16(
            false, a[mt].v, false, bf[nt].v, (short)0, acc[mt][nt], false, false);
  }

  const long outBase = (long)bb * Odim * NTOT;
  #pragma unroll
  for (int mt = 0; mt < 2; ++mt)
    #pragma unroll
    for (int nt = 0; nt < 4; ++nt)
      #pragma unroll
      for (int r = 0; r < 8; ++r) {
        int m = bM + wm + mt * 16 + r + 8 * hi;
        int n = bN + wn + nt * 16 + l16;
        float v = acc[mt][nt][r] * scale;
        if (bias)  v += bias[m];
        if (resid) v += resid[outBase + (long)m * NTOT + n];
        if (outf)  outf[outBase + (long)m * NTOT + n] = v;
        else       outb[outBase + (long)m * NTOT + n] = f2bf(v);
      }
}

// ---------------------------------------------------------------------------
// Flash attention: grid (N/128, NH=4, B). 8 waves/block, 16 queries per wave.
// Q pre-scaled by 1/sqrt(C). Online softmax over 4096 keys in chunks of 32.
// V tile staged into LDS by the Tensor Data Mover (wave 0), K tile staged
// transposed by all threads.
// ---------------------------------------------------------------------------
__global__ void __launch_bounds__(256)
attn_kernel(const unsigned short* __restrict__ Q,    // [B,256,N]
            const unsigned short* __restrict__ KV,   // [B,512,N]
            unsigned short* __restrict__ O)          // [B,256,N]
{
  const int h   = blockIdx.y;
  const int b   = blockIdx.z;
  const int tid = threadIdx.x;
  const int lane = tid & 31;
  const int w    = tid >> 5;
  const int hi   = lane >> 4;
  const int l16  = lane & 15;
  const int q0   = blockIdx.x * 128 + w * 16;

  const unsigned short* qb = Q  + ((long)b * 256 + h * 64) * NTOT;
  const unsigned short* kb = KV + ((long)b * 512 + h * 128) * NTOT;
  const unsigned short* vb = kb + (long)64 * NTOT;

  __shared__ unsigned short lK[32 * 66];     // [key][c]  stride 66
  __shared__ unsigned short lV[64 * 34];     // [c][key]  stride 34 (pad 1 dword/row)
  __shared__ float          sS[8][512];      // per-wave [q16][key32]
  __shared__ unsigned short sP[8][16 * 34];  // per-wave P, stride 34
  __shared__ float          sA[8][16];       // per-wave alpha / inv_l
  unsigned* lKu = (unsigned*)lK;
  unsigned* lVu = (unsigned*)lV;
  unsigned* sPu = (unsigned*)sP[w];

  // Q A-fragments (M=query, K=channel), two 32-channel steps, loaded once
  Frag16 qf[2];
  #pragma unroll
  for (int cs = 0; cs < 2; ++cs)
    #pragma unroll
    for (int i = 0; i < 16; ++i) {
      int c = cs * 32 + i + 16 * hi;
      qf[cs].h[i] = qb[(long)c * NTOT + q0 + l16];
    }

  v8f zero = {0.f, 0.f, 0.f, 0.f, 0.f, 0.f, 0.f, 0.f};
  v8f oac[4];
  #pragma unroll
  for (int t = 0; t < 4; ++t) oac[t] = zero;
  float m_i = -3.0e38f, l_i = 0.f;   // valid in lanes 0..15 (query = q0+lane)

  for (int kc = 0; kc < NTOT; kc += 32) {
    __syncthreads();
#if USE_TDM
    if (w == 0) {
      // TDM descriptor: 2-D tile, 32 keys (dim0, 2B elems) x 64 channel rows,
      // tensor row stride 4096 elems; LDS pad 1 dword after every 16 dwords
      // (-> padded row stride 34 u16 matching lV).
      unsigned long long ga = (unsigned long long)(const void*)(vb + kc);
      unsigned ldsOff = (unsigned)(unsigned long long)(const void*)&lV[0];
      u32x4 g0 = { 1u,                                     // count=1, user D#
                   ldsOff,                                 // lds_addr
                   (unsigned)ga,                           // global_addr[31:0]
                   ((unsigned)(ga >> 32) & 0x01FFFFFFu) | (2u << 30) }; // [56:32] | type=2
      i32x8 g1 = { (int)((1u << 16) | (1u << 20) | (3u << 22)), // data_size=2B, pad_en, pad_interval=16dw, pad_amount=1dw
                   (int)(4096u << 16),                     // tensor_dim0 = 4096
                   (int)(64u << 16),                       // tensor_dim1 = 64
                   (int)(32u << 16),                       // tile_dim0   = 32
                   64,                                     // tile_dim1   = 64
                   4096, 0, 0 };                           // tensor_dim0_stride = 4096
      i32x4 gz4 = {0, 0, 0, 0};
#if __clang_major__ >= 23
      i32x8 gz8 = {0, 0, 0, 0, 0, 0, 0, 0};
      __builtin_amdgcn_tensor_load_to_lds(g0, g1, gz4, gz4, gz8, 0);
#else
      __builtin_amdgcn_tensor_load_to_lds(g0, g1, gz4, gz4, 0);
#endif
    }
#endif
    {   // cooperative stage of K tile (transposed -> lK[key][c])
      int c = tid >> 2, seg = (tid & 3) * 8;
      const unsigned short* kp = kb + (long)c * NTOT + kc + seg;
      #pragma unroll
      for (int j = 0; j < 8; ++j) lK[(seg + j) * 66 + c] = kp[j];
      if (kc + 32 < NTOT) __builtin_prefetch(kp + 32, 0, 3);
#if !USE_TDM
      const unsigned short* vp = vb + (long)c * NTOT + kc + seg;
      #pragma unroll
      for (int j = 0; j < 8; ++j) lV[c * 34 + seg + j] = vp[j];
#endif
    }
#if USE_TDM
    if (w == 0) __builtin_amdgcn_s_wait_tensorcnt(0);
#endif
    __syncthreads();

    // S(16x32) = Q^T K  (2 key sub-tiles, accumulate over 2 channel steps)
    v8f s0 = zero, s1 = zero;
    #pragma unroll
    for (int cs = 0; cs < 2; ++cs) {
      Frag16 bk0, bk1;
      #pragma unroll
      for (int j = 0; j < 8; ++j) {
        bk0.u[j] = lKu[(l16)      * 33 + cs * 16 + j + 8 * hi];
        bk1.u[j] = lKu[(16 + l16) * 33 + cs * 16 + j + 8 * hi];
      }
      s0 = __builtin_amdgcn_wmma_f32_16x16x32_bf16(false, qf[cs].v, false, bk0.v, (short)0, s0, false, false);
      s1 = __builtin_amdgcn_wmma_f32_16x16x32_bf16(false, qf[cs].v, false, bk1.v, (short)0, s1, false, false);
    }

    // spill S to per-wave LDS (D layout: M = r + 8*hi, N = l16)
    float* ss = sS[w];
    #pragma unroll
    for (int r = 0; r < 8; ++r) {
      ss[(r + 8 * hi) * 32 + l16]      = s0[r];
      ss[(r + 8 * hi) * 32 + 16 + l16] = s1[r];
    }
    __builtin_amdgcn_wave_barrier();

    if (lane < 16) {   // online softmax: lane owns query row `lane`
      float mx = m_i;
      float row[32];
      #pragma unroll
      for (int j = 0; j < 32; ++j) { row[j] = ss[lane * 32 + j]; mx = fmaxf(mx, row[j]); }
      float alpha = exp2f((m_i - mx) * 1.44269504f);
      float sum = 0.f;
      unsigned short* pp = sP[w] + lane * 34;
      #pragma unroll
      for (int j = 0; j < 32; ++j) {
        float p = exp2f((row[j] - mx) * 1.44269504f);
        sum += p;
        pp[j] = f2bf(p);
      }
      l_i = l_i * alpha + sum;
      m_i = mx;
      sA[w][lane] = alpha;
    }
    __builtin_amdgcn_wave_barrier();

    // rescale O accumulators by alpha of each row
    #pragma unroll
    for (int r = 0; r < 8; ++r) {
      float al = sA[w][r + 8 * hi];
      #pragma unroll
      for (int t = 0; t < 4; ++t) oac[t][r] *= al;
    }

    // P A-fragment
    Frag16 pf;
    #pragma unroll
    for (int j = 0; j < 8; ++j)
      pf.u[j] = sPu[l16 * 17 + j + 8 * hi];

    // O(16x64) += P(16x32) x V^T(32x64)
    #pragma unroll
    for (int ct = 0; ct < 4; ++ct) {
      Frag16 bv;
      int c = ct * 16 + l16;
      #pragma unroll
      for (int j = 0; j < 8; ++j)
        bv.u[j] = lVu[c * 17 + j + 8 * hi];
      oac[ct] = __builtin_amdgcn_wmma_f32_16x16x32_bf16(
          false, pf.v, false, bv.v, (short)0, oac[ct], false, false);
    }
  }

  // epilogue: divide by softmax sum, store bf16 (16B packed stores)
  if (lane < 16) sA[w][lane] = 1.0f / l_i;
  __builtin_amdgcn_wave_barrier();
  float inv[8];
  #pragma unroll
  for (int r = 0; r < 8; ++r) inv[r] = sA[w][r + 8 * hi];

  unsigned short* ob = O + ((long)b * 256 + h * 64) * NTOT;
  #pragma unroll
  for (int ct = 0; ct < 4; ++ct) {
    int c = ct * 16 + l16;
    unsigned pk[4];
    #pragma unroll
    for (int j = 0; j < 4; ++j) {
      unsigned short h0 = f2bf(oac[ct][2 * j]     * inv[2 * j]);
      unsigned short h1 = f2bf(oac[ct][2 * j + 1] * inv[2 * j + 1]);
      pk[j] = (unsigned)h0 | ((unsigned)h1 << 16);
    }
    *(uint4*)(ob + (long)c * NTOT + q0 + 8 * hi) = *(uint4*)pk;
  }
}

// ---------------------------------------------------------------------------
extern "C" void kernel_launch(void* const* d_in, const int* in_sizes, int n_in,
                              void* d_out, int out_size, void* d_ws, size_t ws_size,
                              hipStream_t stream)
{
  const float* input  = (const float*)d_in[0];
  const float* cctx   = (const float*)d_in[1];
  const float* gn_w   = (const float*)d_in[2];
  const float* gn_b   = (const float*)d_in[3];
  const float* wq     = (const float*)d_in[4];
  const float* wkv    = (const float*)d_in[5];
  const float* wout_w = (const float*)d_in[6];
  const float* wout_b = (const float*)d_in[7];
  float* out = (float*)d_out;

  const long SZ = 2L * 256 * 4096;                 // 2,097,152 bf16 elems
  unsigned short* ws  = (unsigned short*)d_ws;     // needs 6*SZ*2 = 24 MiB
  unsigned short* nq  = ws;
  unsigned short* nkv = ws + SZ;
  unsigned short* qb  = ws + 2 * SZ;
  unsigned short* kvb = ws + 3 * SZ;               // 2*SZ long
  unsigned short* ob  = ws + 5 * SZ;

  gn_kernel<<<dim3(64, 2, 1), 256, 0, stream>>>(input, cctx, gn_w, gn_b, nq, nkv);
  // q = (1/sqrt(256)) * wq @ nq
  gemm_bf16<<<dim3(32, 2, 2), 256, 0, stream>>>(wq, nq, qb, nullptr, nullptr, nullptr, 256, 0.0625f);
  // kv = wkv @ nkv
  gemm_bf16<<<dim3(32, 4, 2), 256, 0, stream>>>(wkv, nkv, kvb, nullptr, nullptr, nullptr, 512, 1.0f);
  // attention
  attn_kernel<<<dim3(32, 4, 2), 256, 0, stream>>>(qb, kvb, ob);
  // out = wout @ o + bias + input   (fp32)
  gemm_bf16<<<dim3(32, 2, 2), 256, 0, stream>>>(wout_w, ob, nullptr, out, wout_b, input, 256, 1.0f);
}
